// FlashFFTConvMetal_73615739453549
// MI455X (gfx1250) — compile-verified, matrix operand order checked by reading, not run on hardware
//
#include <hip/hip_runtime.h>

#define BATCH    8
#define CHANNELS 768
#define SEQLEN   4096

#define WAVES    4       // waves per workgroup
#define TPW      8       // output tiles per wave (acc=64 VGPR, A=64 VGPR)
#define GROUPS   8       // workgroups per channel; TPW*WAVES*GROUPS == 256 tiles

#define KREV_LEN 4160    // 4096 + zero padding for the causal triangle
#define NPHASE   4       // phase-shifted reversed-k copies -> aligned ds_load_b64

typedef __attribute__((ext_vector_type(16))) _Float16 v16h;
typedef __attribute__((ext_vector_type(8)))  _Float16 v8h;
typedef __attribute__((ext_vector_type(4)))  _Float16 v4h;
typedef __attribute__((ext_vector_type(8)))  float    v8f;

// Causal conv as block-Toeplitz GEMM:
//   D[16 outs x 16 cols(8 batches)] += A(k-Toeplitz 16x32) x B(u 32x16), f32 acc.
// u chunks staged global->LDS with CDNA5 async-DMA (ASYNCcnt), k staged once.
__global__ __launch_bounds__(WAVES * 32)
void flashconv_toeplitz_wmma(const float* __restrict__ u,
                             const float* __restrict__ k,
                             const float* __restrict__ Dvec,
                             const float* __restrict__ mask,
                             float* __restrict__ out)
{
    __shared__ __align__(16) _Float16 krev[NPHASE][KREV_LEN];   // ~33 KB
    __shared__ __align__(16) float    ubuf[2][BATCH][32];       // raw f32 u chunk (async dest)

    const int c    = blockIdx.x;                          // channel
    const int g    = blockIdx.y;                          // tile group
    const int tid  = threadIdx.x;
    const int lane = tid & 31;
    const int wave = __builtin_amdgcn_readfirstlane(tid >> 5);  // scalar wave id

    const float* kc = k + (size_t)c * SEQLEN;

    // ---- stage reversed k once (unconditional clamped load + select, no branches)
    for (int i = tid; i < KREV_LEN; i += WAVES * 32) {
        int src  = SEQLEN - 1 - i;
        int srcc = src < 0 ? 0 : src;
        float v  = kc[srcc];
        krev[0][i] = (_Float16)((src >= 0) ? v : 0.0f);
    }
    __syncthreads();
    for (int p = 1; p < NPHASE; ++p)
        for (int i = tid; i < KREV_LEN; i += WAVES * 32)
            krev[p][i] = (i >= p) ? krev[0][i - p] : (_Float16)0.0f;
    __syncthreads();

    v8f  acc[TPW]  = {};
    v16h Areg[TPW] = {};

    const int hi    = (lane >= 16) ? 8 : 0;   // K-half select (16-bit WMMA layout)
    const int r     = lane & 15;              // A row (M) / B column (N)
    const int base  = wave * 128 + g * 16;    // n0t = 512*tt + base  (scalar)
    const int n0max = (248 + g) << 4;         // max n0 over this workgroup
    const v16h zero16 = {};

    int par = 0;
    for (int j0 = 0; j0 <= n0max; j0 += 32, par ^= 1) {
        // ---- async DMA u[b, c, j0..j0+31] into LDS: 2x b128 per staging lane
        if (tid < 32) {
            const int b  = tid >> 2;              // 0..7
            const int s0 = (tid & 3) << 3;        // 0,8,16,24
            const float* up = u + ((size_t)b * CHANNELS + c) * SEQLEN + j0 + s0;
            // generic->LDS byte offset: ISA 10.2, low 32 bits of the address
            unsigned l0 = (unsigned)(unsigned long long)&ubuf[par][b][s0];
            unsigned l1 = l0 + 16;
            asm volatile(
                "global_load_async_to_lds_b128 %0, %2, off\n\t"
                "global_load_async_to_lds_b128 %1, %3, off"
                :: "v"(l0), "v"(l1), "v"(up), "v"(up + 4)
                : "memory");
            if (j0 + 32 <= n0max)
                __builtin_prefetch(up + 32, 0, 1);      // global_prefetch_b8
        }
        asm volatile("s_wait_asynccnt 0x0" ::: "memory");  // DMA -> LDS visible
        __syncthreads();   // single barrier per chunk (double-buffered ubuf)

        // ---- B operand: 4x aligned b128 LDS loads + f32->f16 converts
        const int rb = (r < BATCH) ? r : 0;
        const float* ub = &ubuf[par][rb][0];
        float4 f0 = *(const float4*)&ub[hi];        // K = hi..hi+3
        float4 f1 = *(const float4*)&ub[hi + 4];    // K = hi+4..hi+7
        float4 f2 = *(const float4*)&ub[hi + 16];   // K = hi+16..hi+19
        float4 f3 = *(const float4*)&ub[hi + 20];   // K = hi+20..hi+23
        v16h bmat;
        bmat[0]  = (_Float16)f0.x; bmat[1]  = (_Float16)f0.y;
        bmat[2]  = (_Float16)f0.z; bmat[3]  = (_Float16)f0.w;
        bmat[4]  = (_Float16)f1.x; bmat[5]  = (_Float16)f1.y;
        bmat[6]  = (_Float16)f1.z; bmat[7]  = (_Float16)f1.w;
        bmat[8]  = (_Float16)f2.x; bmat[9]  = (_Float16)f2.y;
        bmat[10] = (_Float16)f2.z; bmat[11] = (_Float16)f2.w;
        bmat[12] = (_Float16)f3.x; bmat[13] = (_Float16)f3.y;
        bmat[14] = (_Float16)f3.z; bmat[15] = (_Float16)f3.w;
        if (r >= BATCH) bmat = zero16;

        // ---- phase 1: issue ALL A-operand ds loads (per-tile register sets)
        #pragma unroll
        for (int tt = 0; tt < TPW; ++tt) {
            const int n0t = tt * 512 + base;
            if (j0 <= n0t) {                                  // scalar causal skip
                const int jr1 = (SEQLEN - 1) - n0t + j0 - r + hi;
                const int p   = (NPHASE - (jr1 & (NPHASE - 1))) & (NPHASE - 1);
                const int off = jr1 + p;                      // multiple of 4 halves
                const v4h* kb = (const v4h*)&krev[p][off];    // 8B aligned
                v4h q0 = kb[0], q1 = kb[1];                   // K = hi..hi+7
                v4h q2 = kb[4], q3 = kb[5];                   // K = hi+16..hi+23
                v8h alo = __builtin_shufflevector(q0, q1, 0,1,2,3,4,5,6,7);
                v8h ahi = __builtin_shufflevector(q2, q3, 0,1,2,3,4,5,6,7);
                Areg[tt] = __builtin_shufflevector(alo, ahi,
                               0,1,2,3,4,5,6,7,8,9,10,11,12,13,14,15);
            }
        }
        // ---- phase 2: WMMA chain (loads already in flight; graduated waits)
        #pragma unroll
        for (int tt = 0; tt < TPW; ++tt) {
            const int n0t = tt * 512 + base;
            if (j0 <= n0t)
                acc[tt] = __builtin_amdgcn_wmma_f32_16x16x32_f16(
                              false, Areg[tt], false, bmat,
                              (short)0, acc[tt], false, false);
        }
    }

    // ---- epilogue: residual D, exact-erf GELU, dropout mask; 32B-contig stores
    const int bcol = lane & 15;
    if (bcol < BATCH) {
        const float Dc = Dvec[c];
        const float mm = mask[bcol * CHANNELS + c];
        const float* ubc = u   + ((size_t)bcol * CHANNELS + c) * SEQLEN;
        float*       obc = out + ((size_t)bcol * CHANNELS + c) * SEQLEN;
        const int mofs = (lane >= 16) ? 8 : 0;
        #pragma unroll
        for (int tt = 0; tt < TPW; ++tt) {
            const int n0t = tt * 512 + base;
            #pragma unroll
            for (int v = 0; v < 8; ++v) {
                const int n = n0t + v + mofs;
                float y = acc[tt][v] + ubc[n] * Dc;
                y = 0.5f * y * (1.0f + erff(y * 0.70710678118654752f));
                obc[n] = y * mm;
            }
        }
    }
}

extern "C" void kernel_launch(void* const* d_in, const int* in_sizes, int n_in,
                              void* d_out, int out_size, void* d_ws, size_t ws_size,
                              hipStream_t stream)
{
    (void)in_sizes; (void)n_in; (void)out_size; (void)d_ws; (void)ws_size;
    const float* u    = (const float*)d_in[0];   // [8, 768, 4096]
    const float* k    = (const float*)d_in[1];   // [768, 4096]
    const float* Dv   = (const float*)d_in[2];   // [768]
    const float* mask = (const float*)d_in[3];   // [8, 768]
    float*       out  = (float*)d_out;           // [8, 768, 4096]

    dim3 grid(CHANNELS, GROUPS);                 // (768, 8)
    dim3 block(WAVES * 32);                      // 4 wave32
    flashconv_toeplitz_wmma<<<grid, block, 0, stream>>>(u, k, Dv, mask, out);
}
